// SeqEncoder_51659866636623
// MI455X (gfx1250) — matrix operand order repaired and checked by reading, not compiled, not run on hardware
//
#include <hip/hip_runtime.h>
#include <hip/hip_bf16.h>

typedef __attribute__((ext_vector_type(16))) _Float16 v16h;
typedef __attribute__((ext_vector_type(8)))  _Float16 v8h;
typedef __attribute__((ext_vector_type(8)))  float    v8f;
typedef unsigned int u32x4 __attribute__((ext_vector_type(4)));
typedef int          i32x4 __attribute__((ext_vector_type(4)));
typedef int          i32x8 __attribute__((ext_vector_type(8)));

#define B_  32
#define S_  512
#define D_  512
#define H_  8
#define DH_ 64

#if defined(__has_include)
#  if __has_include(<hip/amd_detail/amd_gfx1250_TDM.h>)
#    define TDM_SIX_ARG 1
#  endif
#endif

// ---- WMMA helpers ------------------------------------------------------
__device__ __forceinline__ v8f wmma_f16(v16h a, v16h b, v8f c) {
    // (neg_a, A, neg_b, B, c_mod, C, reuse_a, reuse_b)
    return __builtin_amdgcn_wmma_f32_16x16x32_f16(false, a, false, b,
                                                  (short)0, c, false, false);
}

// A-matrix 16x32 f16 tile, row-major source. lane = M (mod 16).
// elems 0..7 -> K = kbase + khalf + 0..7 ; elems 8..15 -> K = kbase+16+khalf+0..7
__device__ __forceinline__ v16h load_a_tile(const _Float16* rowbase, int kbase, int lane) {
    int khalf = ((lane >> 4) << 3);                    // 0 or 8
    v8h lo = *(const v8h*)(rowbase + kbase + khalf);
    v8h hi = *(const v8h*)(rowbase + kbase + khalf + 16);
    v16h r;
#pragma unroll
    for (int i = 0; i < 8; ++i) { r[i] = lo[i]; r[i + 8] = hi[i]; }
    return r;
}

// B-matrix 32x16 f16 tile, loaded from B^T rows (N rows of length K). lane = N.
// elems 0..7 -> K = kbase + khalf + 0..7 ; elems 8..15 -> K = kbase+16+khalf+0..7
__device__ __forceinline__ v16h load_b_tile(const _Float16* rowbase, int kbase, int lane) {
    int koff = ((lane >> 4) << 3);                     // 0 or 8
    v8h lo = *(const v8h*)(rowbase + kbase + koff);
    v8h hi = *(const v8h*)(rowbase + kbase + koff + 16);
    v16h r;
#pragma unroll
    for (int i = 0; i < 8; ++i) { r[i] = lo[i]; r[i + 8] = hi[i]; }
    return r;
}

// ---- TDM: 2D tile Global -> LDS (D# per CDNA5 ISA) ----
// dims/strides in elements; data_size = 2 bytes (f16)
__device__ __forceinline__ void tdm_load_2d_f16_to_lds(unsigned int lds_off,
                                                       const void* gptr,
                                                       unsigned int tensor_d0,
                                                       unsigned int tensor_d1,
                                                       unsigned int tile_d0,
                                                       unsigned int tile_d1,
                                                       unsigned int stride0) {
    unsigned long long ga = (unsigned long long)(uintptr_t)gptr;
    u32x4 g0;
    g0[0] = 1u;                                            // count=1, user descriptor
    g0[1] = lds_off;                                       // LDS byte address
    g0[2] = (unsigned int)ga;                              // global addr [31:0]
    g0[3] = (unsigned int)((ga >> 32) & 0x01FFFFFFu)       // global addr [56:32]
          | (2u << 30);                                    // type = 2 ("image")
    i32x8 g1;
    g1[0] = (int)(1u << 16);                               // wg_mask=0, data_size=1 (2B)
    g1[1] = (int)((tensor_d0 & 0xFFFFu) << 16);            // tensor_dim0[15:0]
    g1[2] = (int)((tensor_d0 >> 16) | ((tensor_d1 & 0xFFFFu) << 16));
    g1[3] = (int)((tensor_d1 >> 16) | ((tile_d0 & 0xFFFFu) << 16));
    g1[4] = (int)(tile_d1 & 0xFFFFu);                      // tile_dim1 (tile_dim2 = 0)
    g1[5] = (int)stride0;                                  // tensor_dim0_stride[31:0]
    g1[6] = 0;
    g1[7] = 0;
    i32x4 gz = {0, 0, 0, 0};
#ifdef TDM_SIX_ARG
    i32x8 gz8 = {0, 0, 0, 0, 0, 0, 0, 0};
    __builtin_amdgcn_tensor_load_to_lds(g0, g1, gz, gz, gz8, 0);
#else
    __builtin_amdgcn_tensor_load_to_lds(g0, g1, gz, gz, 0);
#endif
}

// ---- Kernel 1: embedding + LayerNorm -> f16 ---------------------------
// one wave per token; 4 waves per block
__global__ void embed_ln_kernel(const int* __restrict__ x_seq,
                                const float* __restrict__ word_emb,
                                const float* __restrict__ pos_emb,
                                const float* __restrict__ ln_w,
                                const float* __restrict__ ln_b,
                                _Float16* __restrict__ h16) {
    int lane = threadIdx.x & 31;
    int wave = threadIdx.x >> 5;
    int t = blockIdx.x * 4 + wave;                     // token index in [0, B*S)
    int s = t & (S_ - 1);
    int tok = x_seq[t];
    const float* we = word_emb + (size_t)tok * D_;
    const float* pe = pos_emb + (size_t)s * D_;
    float e[16];
    float sum = 0.f, sq = 0.f;
#pragma unroll
    for (int i = 0; i < 16; ++i) {
        int c = lane + i * 32;
        float v = we[c] + pe[c];
        e[i] = v; sum += v; sq += v * v;
    }
#pragma unroll
    for (int off = 16; off > 0; off >>= 1) {
        sum += __shfl_xor(sum, off);
        sq  += __shfl_xor(sq,  off);
    }
    float mu   = sum * (1.0f / D_);
    float var  = sq  * (1.0f / D_) - mu * mu;          // biased variance
    float rstd = rsqrtf(var + 1e-5f);
    _Float16* out = h16 + (size_t)t * D_;
#pragma unroll
    for (int i = 0; i < 16; ++i) {
        int c = lane + i * 32;
        out[c] = (_Float16)((e[i] - mu) * rstd * ln_w[c] + ln_b[c]);
    }
}

// ---- Kernel 2: weight f32 -> f16 --------------------------------------
__global__ void wcvt_kernel(const float* __restrict__ wq,
                            const float* __restrict__ wk,
                            const float* __restrict__ wv,
                            _Float16* __restrict__ w16) {
    int i = blockIdx.x * blockDim.x + threadIdx.x;     // < 3*D*D
    const int DD = D_ * D_;
    const float* src = (i < DD) ? wq : (i < 2 * DD) ? wk : wv;
    int j = (i < DD) ? i : (i < 2 * DD) ? (i - DD) : (i - 2 * DD);
    w16[i] = (_Float16)src[j];
}

// ---- Kernel 3: QKV projection (x @ W^T + b) via WMMA ------------------
// grid (B*S/16, 3); block 128 (4 waves). wave w: 16 rows x cols [w*128, w*128+128)
// A strip (16 tokens x 512) staged once into LDS by the Tensor Data Mover;
// all 4 waves consume it via ds_load.
__global__ void qkv_kernel(const _Float16* __restrict__ h16,
                           const _Float16* __restrict__ w16,   // [3][D][D]
                           const float* __restrict__ bq,
                           const float* __restrict__ bk,
                           const float* __restrict__ bv,
                           _Float16* __restrict__ q16,   // [B,H,S,DH]
                           _Float16* __restrict__ k16,   // [B,H,S,DH]
                           _Float16* __restrict__ vT16)  // [B,H,DH,S]
{
    __shared__ _Float16 lds_a[16 * D_];                 // 16 KB
    int lane = threadIdx.x & 31;
    int wave = threadIdx.x >> 5;
    int row0 = blockIdx.x * 16;
    int wgt  = blockIdx.y;                              // 0=q 1=k 2=v
    const _Float16* W    = w16 + (size_t)wgt * D_ * D_;
    const float*    bias = (wgt == 0) ? bq : (wgt == 1) ? bk : bv;
    int m15 = lane & 15;
    int ncol0 = wave * 128;

    if (wave == 0) {
        tdm_load_2d_f16_to_lds((unsigned int)(uintptr_t)lds_a,
                               h16 + (size_t)row0 * D_,
                               /*tensor_d0=*/D_, /*tensor_d1=*/16,
                               /*tile_d0=*/D_,   /*tile_d1=*/16,
                               /*stride0=*/D_);
        __builtin_amdgcn_s_wait_tensorcnt(0);
    }
    __syncthreads();

    const _Float16* arow = lds_a + (size_t)m15 * D_;

    v8f acc[8];
#pragma unroll
    for (int n = 0; n < 8; ++n) acc[n] = {};

    for (int kb = 0; kb < D_; kb += 32) {
        v16h a = load_a_tile(arow, kb, lane);           // ds_load_b128 x2
#pragma unroll
        for (int n = 0; n < 8; ++n) {
            const _Float16* brow = W + (size_t)(ncol0 + n * 16 + m15) * D_;
            v16h bt = load_b_tile(brow, kb, lane);
            acc[n] = wmma_f16(a, bt, acc[n]);
        }
    }

    int moff = (lane >> 4) << 3;                        // 0 or 8
#pragma unroll
    for (int n = 0; n < 8; ++n) {
        int c  = ncol0 + n * 16 + m15;
        int hh = c >> 6, dh = c & 63;
        float bsv = bias[c];
#pragma unroll
        for (int r = 0; r < 8; ++r) {
            int t = row0 + r + moff;                    // token index
            int b = t >> 9, s = t & (S_ - 1);
            _Float16 val = (_Float16)(acc[n][r] + bsv);
            if (wgt == 0)
                q16[(((size_t)(b * H_ + hh)) * S_ + s) * DH_ + dh] = val;
            else if (wgt == 1)
                k16[(((size_t)(b * H_ + hh)) * S_ + s) * DH_ + dh] = val;
            else
                vT16[(((size_t)(b * H_ + hh)) * DH_ + dh) * S_ + s] = val;
        }
    }
}

// ---- Kernel 4: attention (scores -> mask -> softmax -> probs, ctx) ----
// grid B*H*S/32; block 64 (2 waves); each wave owns 16 query rows.
// dynamic LDS: 2 * 16 * 512 * 4 = 64 KB
__global__ void attn_kernel(const _Float16* __restrict__ q16,
                            const _Float16* __restrict__ k16,
                            const _Float16* __restrict__ vT16,
                            const int* __restrict__ mask,
                            float* __restrict__ out0,     // [B,D]
                            float* __restrict__ ctx,      // [B,S,D]
                            float* __restrict__ probs)    // [B,H,S,S]
{
    extern __shared__ float smem[];
    int lane = threadIdx.x & 31;
    int wave = threadIdx.x >> 5;
    int bh   = blockIdx.x >> 4;                         // b*H + h
    int qblk = blockIdx.x & 15;
    int b    = bh >> 3;
    int hh   = bh & 7;
    int q0   = qblk * 32 + wave * 16;
    int m15  = lane & 15;
    int moff = (lane >> 4) << 3;
    float* sm = smem + (size_t)wave * (16 * 512);

    // ---- scores = (q . k / 8) * ((1-mask)*-1e4), staged in LDS --------
    const _Float16* qrow = q16 + ((size_t)bh * S_ + q0 + m15) * DH_;
    v16h a0 = load_a_tile(qrow, 0, lane);
    v16h a1 = load_a_tile(qrow, 32, lane);
    for (int kt = 0; kt < 32; ++kt) {
        int key0 = kt * 16;
        const _Float16* krow = k16 + ((size_t)bh * S_ + key0 + m15) * DH_;
        v16h b0 = load_b_tile(krow, 0, lane);
        v16h b1 = load_b_tile(krow, 32, lane);
        v8f acc = {};
        acc = wmma_f16(a0, b0, acc);
        acc = wmma_f16(a1, b1, acc);
        float mv    = (float)mask[b * S_ + key0 + m15]; // one column per lane
        float scale = 0.125f * ((1.0f - mv) * -10000.0f);
#pragma unroll
        for (int r = 0; r < 8; ++r)
            sm[(r + moff) * 512 + key0 + m15] = acc[r] * scale;
    }
    __syncthreads();

    // ---- stable softmax over 512 keys; 2 lanes per row ----------------
    {
        int row  = lane >> 1;
        int half = lane & 1;
        float* rp = sm + row * 512 + half * 256;
        float mx = -3.4e38f;
        for (int i = 0; i < 256; ++i) mx = fmaxf(mx, rp[i]);
        mx = fmaxf(mx, __shfl_xor(mx, 1));
        float ssum = 0.f;
        for (int i = 0; i < 256; ++i) {
            float e = __expf(rp[i] - mx);
            rp[i] = e;
            ssum += e;
        }
        ssum += __shfl_xor(ssum, 1);
        float inv = 1.0f / ssum;
        for (int i = 0; i < 256; ++i) rp[i] *= inv;
    }
    __syncthreads();

    // ---- stream probs tile out (float4) -------------------------------
    {
        float4*       dst = (float4*)(probs + ((size_t)bh * S_ + q0) * S_);
        const float4* src = (const float4*)sm;
        for (int i = lane; i < 16 * 128; i += 32) dst[i] = src[i];
    }

    // ---- ctx = probs @ v via WMMA (A from LDS f32->f16, B from vT) ----
    v8f cacc[4];
#pragma unroll
    for (int n = 0; n < 4; ++n) cacc[n] = {};
    const float* prow = sm + (size_t)m15 * 512;
    int khalf = (lane >> 4) << 3;
    for (int kb = 0; kb < 512; kb += 32) {
        const float* p0 = prow + kb + khalf;
        const float* p1 = p0 + 16;
        v16h a;
#pragma unroll
        for (int i = 0; i < 8; ++i) {
            a[i]     = (_Float16)p0[i];
            a[i + 8] = (_Float16)p1[i];
        }
#pragma unroll
        for (int n = 0; n < 4; ++n) {
            const _Float16* vrow = vT16 + ((size_t)bh * DH_ + n * 16 + m15) * S_;
            v16h bt = load_b_tile(vrow, kb, lane);
            cacc[n] = wmma_f16(a, bt, cacc[n]);
        }
    }
#pragma unroll
    for (int n = 0; n < 4; ++n) {
#pragma unroll
        for (int r = 0; r < 8; ++r) {
            int qi  = q0 + r + moff;
            int col = hh * 64 + n * 16 + m15;
            float val = cacc[n][r];
            ctx[((size_t)b * S_ + qi) * D_ + col] = val;
            if (qi == 0) out0[b * D_ + col] = val;
        }
    }
}

// ---- host launch -------------------------------------------------------
extern "C" void kernel_launch(void* const* d_in, const int* in_sizes, int n_in,
                              void* d_out, int out_size, void* d_ws, size_t ws_size,
                              hipStream_t stream) {
    const int*   x_seq    = (const int*)d_in[0];
    const int*   mask     = (const int*)d_in[1];
    const float* word_emb = (const float*)d_in[2];
    const float* pos_emb  = (const float*)d_in[3];
    const float* ln_w     = (const float*)d_in[4];
    const float* ln_b     = (const float*)d_in[5];
    const float* wq       = (const float*)d_in[6];
    const float* bq       = (const float*)d_in[7];
    const float* wk       = (const float*)d_in[8];
    const float* bk       = (const float*)d_in[9];
    const float* wv       = (const float*)d_in[10];
    const float* bv       = (const float*)d_in[11];

    float* out   = (float*)d_out;
    float* out0  = out;                                  // [B,D]
    float* ctx   = out + (size_t)B_ * D_;                // [B,S,D]
    float* probs = ctx + (size_t)B_ * S_ * D_;           // [B,H,S,S]

    char* ws = (char*)d_ws;
    _Float16* h16  = (_Float16*)ws;  ws += (size_t)B_ * S_ * D_ * 2;
    _Float16* w16  = (_Float16*)ws;  ws += (size_t)3 * D_ * D_ * 2;
    _Float16* q16  = (_Float16*)ws;  ws += (size_t)B_ * S_ * D_ * 2;
    _Float16* k16  = (_Float16*)ws;  ws += (size_t)B_ * S_ * D_ * 2;
    _Float16* vT16 = (_Float16*)ws;  ws += (size_t)B_ * S_ * D_ * 2;

    embed_ln_kernel<<<B_ * S_ / 4, 128, 0, stream>>>(x_seq, word_emb, pos_emb,
                                                     ln_w, ln_b, h16);
    wcvt_kernel<<<3 * D_ * D_ / 256, 256, 0, stream>>>(wq, wk, wv, w16);
    qkv_kernel<<<dim3(B_ * S_ / 16, 3), 128, 0, stream>>>(h16, w16, bq, bk, bv,
                                                          q16, k16, vT16);
    attn_kernel<<<B_ * H_ * S_ / 32, 64, 2 * 16 * 512 * sizeof(float), stream>>>(
        q16, k16, vT16, mask, out0, ctx, probs);
}
